// MultiTaskFEGIN_9088150798767
// MI455X (gfx1250) — matrix operand order repaired and checked by reading, not compiled, not run on hardware
//
#include <hip/hip_runtime.h>
#include <hip/hip_bf16.h>
#include <hip/hip_fp16.h>

typedef __attribute__((ext_vector_type(16))) _Float16 v16h;
typedef __attribute__((ext_vector_type(8)))  float    v8f;
typedef __attribute__((__vector_size__(16))) int      async_i128;  // 16B copy unit

#if defined(__AMDGCN__) && __has_builtin(__builtin_amdgcn_global_load_async_to_lds_b128) && \
    __has_builtin(__builtin_amdgcn_s_wait_asynccnt)
#define USE_ASYNC_LDS 1
#else
#define USE_ASYNC_LDS 0
#endif

// ---------------------------------------------------------------- utilities

__global__ void zero_f32(float* __restrict__ p, long n) {
  long i = (long)blockIdx.x * blockDim.x + threadIdx.x;
  if (i < n) p[i] = 0.0f;
}

// Convert fp32 weights [KDIM x 128] into f16 WMMA B-fragment-major order:
// wf[((ks*8 + ct)*32 + lane)*16 + j] = W[kk*128 + col]
//   kk  = ks*32 + 8*(lane>=16) + j + (j&8)     (zero-padded when kk >= KDIM)
//   col = ct*16 + (lane&15)
template<int KDIM>
__global__ void wconv_frag(const float* __restrict__ w, _Float16* __restrict__ wf) {
  constexpr int KSTEPS = (KDIM + 31) / 32;
  int i = blockIdx.x * blockDim.x + threadIdx.x;
  if (i >= KSTEPS * 4096) return;
  int j    = i & 15;
  int lane = (i >> 4) & 31;
  int ct   = (i >> 9) & 7;
  int ks   = i >> 12;
  int kk   = ks * 32 + ((lane >> 4) << 3) + j + (j & 8);
  int col  = ct * 16 + (lane & 15);
  _Float16 v = (_Float16)0.0f;
  if (kk < KDIM) v = (_Float16)w[(size_t)kk * 128 + col];
  wf[i] = v;
}

// agg[n, :FEAT] = (1+eps) * h[n, :FEAT]   (h row stride = hstride)
template<int FEAT>
__global__ void init_agg(const float* __restrict__ h, int hstride,
                         const float* __restrict__ epsp,
                         float* __restrict__ agg, int N) {
  long i = (long)blockIdx.x * blockDim.x + threadIdx.x;
  if (i >= (long)N * FEAT) return;
  int node = (int)(i / FEAT);
  int f    = (int)(i % FEAT);
  float e = 1.0f + epsp[0];
  agg[(size_t)node * FEAT + f] = e * h[(size_t)node * hstride + f];
}

// agg[dst] += h[src]; LPE lanes per edge, each lane owns 4 consecutive floats.
// Prefetch the gathered row a few thousand edges ahead (global_prefetch_b8).
template<int LPE>
__global__ void scatter_add(const float* __restrict__ h, int hstride,
                            const int* __restrict__ srcI, const int* __restrict__ dstI,
                            float* __restrict__ agg, long E) {
  long tid = (long)blockIdx.x * blockDim.x + threadIdx.x;
  long e = tid / LPE;
  int  l = (int)(tid % LPE);
  if (e >= E) return;
  long e2 = e + 4096;
  if (e2 < E) {
    int s2 = srcI[e2];
    __builtin_prefetch(h + (size_t)s2 * hstride + l * 4, 0, 1);
  }
  int s = srcI[e];
  int d = dstI[e];
  const float4 v = *(const float4*)(h + (size_t)s * hstride + l * 4);
  float* dp = agg + (size_t)d * (LPE * 4) + l * 4;
  atomicAdd(dp + 0, v.x);
  atomicAdd(dp + 1, v.y);
  atomicAdd(dp + 2, v.z);
  atomicAdd(dp + 3, v.w);
}

// ------------------------------------------------------------- WMMA GEMM
// out[N x 128] = relu(A[N x KDIM] @ W + bias), BN stats optionally fused.
// Block = 256 threads = 8 waves. Block owns 128 rows; wave w owns rows
// [w*16, w*16+16) and computes all 8 column tiles (A-fragment reuse = 8).
// A chunks (128 x CW fp32) are staged global->LDS with the gfx1250 async
// copy engine, double-buffered and overlapped with WMMA compute.
// Fragment layouts per CDNA5 ISA 7.12.2.

template<int CW, int CWP>
__device__ __forceinline__ void stage_chunk(float* lds, const float* __restrict__ A,
                                            long row0, int k0, int lda, int N, int tid) {
  constexpr int SEGS = 128 * CW / 4;   // 16-byte segments
  constexpr int PER  = SEGS / 256;
  #pragma unroll
  for (int i = 0; i < PER; ++i) {
    int seg = i * 256 + tid;
    int row = seg / (CW / 4);
    int c0  = (seg % (CW / 4)) * 4;
    long grow = row0 + row;
    if (grow >= N) grow = N - 1;   // clamp; bogus rows masked in epilogue
    const float* gp = A + grow * (long)lda + k0 + c0;
    float* lp = lds + (size_t)row * CWP + c0;
#if USE_ASYNC_LDS
    __builtin_amdgcn_global_load_async_to_lds_b128(
        (__attribute__((address_space(1))) async_i128*)gp,
        (__attribute__((address_space(3))) async_i128*)lp, 0, 0);
#else
    const float4 v = *(const float4*)gp;
    *(float4*)lp = v;
#endif
  }
}

__device__ __forceinline__ void stage_fence() {
#if USE_ASYNC_LDS
  __builtin_amdgcn_s_wait_asynccnt(0);
#endif
  __syncthreads();
}

template<int KDIM, bool STATS>
__global__ void gin_gemm_relu(const float* __restrict__ A, int lda,
                              const _Float16* __restrict__ Wfrag,  // fragment-major
                              const float* __restrict__ bias,
                              float* __restrict__ out,             // [N x 128]
                              int N,
                              float* __restrict__ gsum,
                              float* __restrict__ gsq) {
  constexpr int CW  = (KDIM < 32) ? KDIM : 32;   // chunk width (floats)
  constexpr int CWP = CW + 4;                    // padded LDS row stride
  constexpr int KSTEPS = (KDIM + 31) / 32;
  __shared__ float As[2][128 * CWP];

  const int tid   = threadIdx.x;
  const int lane  = tid & 31;
  const int wave  = tid >> 5;
  const long row0 = (long)blockIdx.x * 128;
  const int rloc  = wave * 16 + (lane & 15);     // local A row for this lane
  const int kbase = (lane >> 4) << 3;            // 0 or 8 (A half-K select)

  v8f acc[8] = {};

  stage_chunk<CW, CWP>(As[0], A, row0, 0, lda, N, tid);
  stage_fence();

  #pragma unroll
  for (int ks = 0; ks < KSTEPS; ++ks) {
    if (ks + 1 < KSTEPS)
      stage_chunk<CW, CWP>(As[(ks + 1) & 1], A, row0, (ks + 1) * 32, lda, N, tid);

    // A fragment from LDS (fp32 -> f16): element j -> K = kbase + j (+8 for j>=8)
    const float* ar = &As[ks & 1][(size_t)rloc * CWP];
    v16h a;
    #pragma unroll
    for (int j = 0; j < 16; ++j) {
      const int kk = kbase + j + (j & 8);
      float av = ((KDIM % 32 == 0) || (kk < CW)) ? ar[kk] : 0.0f;
      a[j] = (_Float16)av;
    }

    // one A fragment feeds 8 column tiles
    #pragma unroll
    for (int ct = 0; ct < 8; ++ct) {
      const v16h b = *(const v16h*)(Wfrag + (((size_t)ks * 8 + ct) << 9) + lane * 16);
      acc[ct] = __builtin_amdgcn_wmma_f32_16x16x32_f16(false, a, false, b,
                                                       (short)0, acc[ct], false, false);
    }

    if (ks + 1 < KSTEPS) stage_fence();
  }

  // epilogue: bias + ReLU + store + per-column BN stats
  // D layout: VGPR p -> row p + 8*(lane>=16), col = lane&15
  const long orow0 = row0 + wave * 16 + ((lane >> 4) << 3);
  if (row0 + 128 <= (long)N) {
    // fast path (all blocks except possibly the last): no row guards
    #pragma unroll
    for (int ct = 0; ct < 8; ++ct) {
      const int bcol = ct * 16 + (lane & 15);
      const float bv = bias[bcol];
      float psum = 0.0f, psq = 0.0f;
      #pragma unroll
      for (int p = 0; p < 8; ++p) {
        float v = acc[ct][p] + bv;
        v = v > 0.0f ? v : 0.0f;
        out[(size_t)(orow0 + p) * 128 + bcol] = v;
        psum += v;
        psq  += v * v;
      }
      if (STATS) {
        atomicAdd(&gsum[bcol], psum);
        atomicAdd(&gsq[bcol],  psq);
      }
    }
  } else {
    // tail block: per-row guards
    #pragma unroll
    for (int ct = 0; ct < 8; ++ct) {
      const int bcol = ct * 16 + (lane & 15);
      const float bv = bias[bcol];
      float psum = 0.0f, psq = 0.0f;
      #pragma unroll
      for (int p = 0; p < 8; ++p) {
        float v = acc[ct][p] + bv;
        v = v > 0.0f ? v : 0.0f;
        if (orow0 + p < N) {
          out[(size_t)(orow0 + p) * 128 + bcol] = v;
          psum += v;
          psq  += v * v;
        }
      }
      if (STATS) {
        atomicAdd(&gsum[bcol], psum);
        atomicAdd(&gsq[bcol],  psq);
      }
    }
  }
}

// ----------------------------------------------------------------- BatchNorm

__global__ void bn_finalize(const float* __restrict__ stats,  // sum[128], sq[128]
                            const float* __restrict__ gamma,
                            const float* __restrict__ beta,
                            float invN,
                            float* __restrict__ scale, float* __restrict__ biasv) {
  int f = threadIdx.x;  // 128 threads
  float mean = stats[f] * invN;
  float var  = stats[128 + f] * invN - mean * mean;
  float sc   = gamma[f] * rsqrtf(var + 1e-5f);
  scale[f] = sc;
  biasv[f] = beta[f] - mean * sc;
}

// Fused: emb[n*512+f] = v = t2[n*128+f]*scale[f]+bias[f];
//        agg[n*128+f] = (1+epsNext)*v   (next layer's aggregation base)
__global__ void bn_apply_init(const float* __restrict__ t2,
                              const float* __restrict__ scale,
                              const float* __restrict__ biasv,
                              float* __restrict__ embcol,   // node_emb + layer*128
                              float* __restrict__ aggOut,   // nullptr for last layer
                              const float* __restrict__ epsNext,
                              int N) {
  long i = (long)blockIdx.x * blockDim.x + threadIdx.x;
  if (i >= (long)N * 128) return;
  int n = (int)(i >> 7);
  int f = (int)(i & 127);
  float v = t2[i] * scale[f] + biasv[f];
  embcol[(size_t)n * 512 + f] = v;
  if (aggOut != nullptr) {
    aggOut[i] = (1.0f + epsNext[0]) * v;
  }
}

// ------------------------------------------------------------------- pooling

__global__ void pool_count(const int* __restrict__ batch, float* __restrict__ counts, int N) {
  int i = blockIdx.x * blockDim.x + threadIdx.x;
  if (i < N) atomicAdd(&counts[batch[i]], 1.0f);
}

__global__ void pool_sum(const float* __restrict__ emb, const int* __restrict__ batch,
                         float* __restrict__ sums, int N) {
  long i = (long)blockIdx.x * blockDim.x + threadIdx.x;
  if (i >= (long)N * 128) return;
  int n  = (int)(i >> 7);
  int c4 = (int)(i & 127);
  int g  = batch[n];
  const float4 v = *(const float4*)(emb + (size_t)n * 512 + c4 * 4);
  float* dp = sums + (size_t)g * 512 + c4 * 4;
  atomicAdd(dp + 0, v.x);
  atomicAdd(dp + 1, v.y);
  atomicAdd(dp + 2, v.z);
  atomicAdd(dp + 3, v.w);
}

// --------------------------------------------------------------- classifier
// 64 blocks (one per graph) x 256 threads; tiny GEMMs done scalar in LDS.
__global__ void classifier(const float* __restrict__ sums, const float* __restrict__ counts,
                           const float* __restrict__ W1, const float* __restrict__ b1,
                           const float* __restrict__ W2, const float* __restrict__ b2,
                           const float* __restrict__ W3, const float* __restrict__ b3,
                           const float* __restrict__ W4, const float* __restrict__ b4,
                           float* __restrict__ out) {
  __shared__ float ge[512];
  __shared__ float h1[256];
  __shared__ float h2[128];
  __shared__ float h3[128];
  __shared__ float lg[10];
  int g = blockIdx.x, t = threadIdx.x;
  float cnt = fmaxf(counts[g], 1.0f);
  for (int c = t; c < 512; c += 256) ge[c] = sums[(size_t)g * 512 + c] / cnt;
  __syncthreads();
  {
    float a = b1[t];
    for (int k = 0; k < 512; ++k) a += ge[k] * W1[k * 256 + t];
    h1[t] = fmaxf(a, 0.0f);
  }
  __syncthreads();
  if (t < 128) {
    float a = b2[t];
    for (int k = 0; k < 256; ++k) a += h1[k] * W2[k * 128 + t];
    h2[t] = fmaxf(a, 0.0f);
  }
  __syncthreads();
  if (t < 128) {
    float a = b3[t];
    for (int k = 0; k < 128; ++k) a += h2[k] * W3[k * 128 + t];
    h3[t] = fmaxf(a, 0.0f);
  }
  __syncthreads();
  if (t < 10) {
    float a = b4[t];
    for (int k = 0; k < 128; ++k) a += h3[k] * W4[k * 10 + t];
    lg[t] = a;
  }
  __syncthreads();
  if (t == 0) {
    float mx = lg[0];
    for (int c = 1; c < 10; ++c) mx = fmaxf(mx, lg[c]);
    float s = 0.0f;
    for (int c = 0; c < 10; ++c) s += expf(lg[c] - mx);
    float lse = mx + logf(s);
    for (int c = 0; c < 10; ++c) out[(size_t)g * 10 + c] = lg[c] - lse;
  }
}

// --------------------------------------------------------------------- host

extern "C" void kernel_launch(void* const* d_in, const int* in_sizes, int n_in,
                              void* d_out, int out_size, void* d_ws, size_t ws_size,
                              hipStream_t stream) {
  (void)n_in; (void)out_size; (void)ws_size;
  const float* x     = (const float*)d_in[0];
  const int*   ei    = (const int*)d_in[1];
  const int*   batch = (const int*)d_in[2];
  const float* c1W1  = (const float*)d_in[3];
  const float* c1b1  = (const float*)d_in[4];
  const float* c1W2  = (const float*)d_in[5];
  const float* c1b2  = (const float*)d_in[6];
  const float* c1g   = (const float*)d_in[7];
  const float* c1be  = (const float*)d_in[8];
  const float* c1eps = (const float*)d_in[9];
  const float* csW1  = (const float*)d_in[10];
  const float* csb1  = (const float*)d_in[11];
  const float* csW2  = (const float*)d_in[12];
  const float* csb2  = (const float*)d_in[13];
  const float* csg   = (const float*)d_in[14];
  const float* csbe  = (const float*)d_in[15];
  const float* cseps = (const float*)d_in[16];
  const float* clW1  = (const float*)d_in[17];
  const float* clb1  = (const float*)d_in[18];
  const float* clW2  = (const float*)d_in[19];
  const float* clb2  = (const float*)d_in[20];
  const float* clW3  = (const float*)d_in[21];
  const float* clb3  = (const float*)d_in[22];
  const float* clW4  = (const float*)d_in[23];
  const float* clb4  = (const float*)d_in[24];

  const int  N = in_sizes[2];        // 100000
  const long E = in_sizes[1] / 2;    // 1600000
  const int  H = 128, F = 16, L = 4;

  char* ws = (char*)d_ws;
  size_t off = 0;
  auto carve = [&](size_t bytes) -> void* {
    void* p = ws + off;
    off += (bytes + 255) & ~(size_t)255;
    return p;
  };
  float*    agg   = (float*)carve((size_t)N * H * 4);
  float*    t1    = (float*)carve((size_t)N * H * 4);
  float*    t2    = (float*)carve((size_t)N * H * 4);
  float*    emb   = (float*)carve((size_t)N * 4 * H * 4);   // [N x 512]
  _Float16* W1f   = (_Float16*)carve(4 * 4096 * 2);         // frag-major, max K=128
  _Float16* W2f   = (_Float16*)carve(4 * 4096 * 2);
  float*    stats = (float*)carve(4 * 128 * 4);             // sum,sq,scale,bias
  float*    pools = (float*)carve((64 * 512 + 64) * 4);     // sums + counts

  const int* srcI = ei;
  const int* dstI = ei + E;
  float* scalev = stats + 256;
  float* biasv  = stats + 384;
  const float invN = 1.0f / (float)N;
  const int GemmBlocks = (N + 127) / 128;   // 782

  for (int layer = 0; layer < L; ++layer) {
    const float* W1 = (layer == 0) ? c1W1 : csW1 + (size_t)(layer - 1) * H * H;
    const float* b1 = (layer == 0) ? c1b1 : csb1 + (size_t)(layer - 1) * H;
    const float* W2 = (layer == 0) ? c1W2 : csW2 + (size_t)(layer - 1) * H * H;
    const float* b2 = (layer == 0) ? c1b2 : csb2 + (size_t)(layer - 1) * H;
    const float* gm = (layer == 0) ? c1g  : csg  + (size_t)(layer - 1) * H;
    const float* bt = (layer == 0) ? c1be : csbe + (size_t)(layer - 1) * H;

    // -------- neighbor aggregation: agg = (1+eps)*h + scatter(h[src] -> dst)
    // (for layer >= 1 the (1+eps)*h base was fused into bn_apply_init)
    if (layer == 0) {
      long n1 = (long)N * F;
      init_agg<16><<<(int)((n1 + 255) / 256), 256, 0, stream>>>(x, F, c1eps, agg, N);
      long th = E * (F / 4);
      scatter_add<4><<<(int)((th + 255) / 256), 256, 0, stream>>>(x, F, srcI, dstI, agg, E);
    } else {
      const float* hprev = emb + (size_t)(layer - 1) * H;   // stride 512
      long th = E * (H / 4);
      scatter_add<32><<<(int)((th + 255) / 256), 256, 0, stream>>>(hprev, 4 * H, srcI, dstI, agg, E);
    }

    // -------- weights -> f16 fragment-major, zero BN stats
    if (layer == 0)
      wconv_frag<16><<<(1 * 4096) / 256, 256, 0, stream>>>(W1, W1f);
    else
      wconv_frag<128><<<(4 * 4096) / 256, 256, 0, stream>>>(W1, W1f);
    wconv_frag<128><<<(4 * 4096) / 256, 256, 0, stream>>>(W2, W2f);
    zero_f32<<<1, 256, 0, stream>>>(stats, 256);

    // -------- MLP: two WMMA GEMMs (+ReLU), BN stats fused into epilogue of #2
    if (layer == 0)
      gin_gemm_relu<16, false><<<GemmBlocks, 256, 0, stream>>>(agg, 16, W1f, b1, t1, N,
                                                               nullptr, nullptr);
    else
      gin_gemm_relu<128, false><<<GemmBlocks, 256, 0, stream>>>(agg, 128, W1f, b1, t1, N,
                                                                nullptr, nullptr);
    gin_gemm_relu<128, true><<<GemmBlocks, 256, 0, stream>>>(t1, 128, W2f, b2, t2, N,
                                                             stats, stats + 128);

    // -------- BatchNorm fold + apply into node_emb slice (+ next agg init)
    bn_finalize<<<1, 128, 0, stream>>>(stats, gm, bt, invN, scalev, biasv);
    const bool hasNext = (layer + 1 < L);
    long n2 = (long)N * H;
    bn_apply_init<<<(int)((n2 + 255) / 256), 256, 0, stream>>>(
        t2, scalev, biasv, emb + (size_t)layer * H,
        hasNext ? agg : nullptr, hasNext ? (cseps + layer) : nullptr, N);
  }

  // -------- global mean pool per graph + classifier
  float* sums   = pools;
  float* counts = pools + 64 * 512;
  zero_f32<<<(64 * 512 + 64 + 255) / 256, 256, 0, stream>>>(pools, 64 * 512 + 64);
  pool_count<<<(N + 255) / 256, 256, 0, stream>>>(batch, counts, N);
  long np = (long)N * 128;
  pool_sum<<<(int)((np + 255) / 256), 256, 0, stream>>>(emb, batch, sums, N);
  classifier<<<64, 256, 0, stream>>>(sums, counts, clW1, clb1, clW2, clb2,
                                     clW3, clb3, clW4, clb4, (float*)d_out);
}
